// SigLipLoss_57346403336666
// MI455X (gfx1250) — compile-verified, matrix operand order checked by reading, not run on hardware
//
#include <hip/hip_runtime.h>
#include <hip/hip_bf16.h>
#include <math.h>

// ---------------------------------------------------------------------------
// SigLIP loss, fused:  loss = (1/N) * sum_ij softplus(-l_ij * (s*<I_i,T_j>+b))
// N = 8192, D = 768, fp32 inputs, scalar fp32 output.
//
// CDNA5 data path:
//   - Tensor Data Mover (tensor_load_to_lds) streams 128x32 fp32 tiles of both
//     operands into double-buffered LDS, overlapped with compute, tracked via
//     TENSORcnt (s_wait_tensorcnt). One barrier per K-step.
//   - v_wmma_f32_16x16x32_bf16 (wave32) with f32 accumulators; fp32->bf16
//     conversion at fragment-load time as one vector fptrunc (backend picks
//     the cvt/pack sequence matching the WMMA operand layout).
//   - Epilogue fused: exp2/log2-based stable softplus, deterministic
//     two-stage reduction (no atomics, no reliance on pre-zeroed buffers).
// ---------------------------------------------------------------------------

typedef __attribute__((ext_vector_type(16))) __bf16        v16bf;
typedef __attribute__((ext_vector_type(16))) float         v16f;
typedef __attribute__((ext_vector_type(8)))  float         v8f;
typedef __attribute__((ext_vector_type(4)))  unsigned int  v4u;
typedef __attribute__((ext_vector_type(8)))  int           v8i;
typedef __attribute__((ext_vector_type(4)))  int           v4i;

#define NROWS 8192
#define DDIM  768
#define BM    128
#define BN    128
#define BK    32
#define SKF   36      // LDS row stride in fp32 elems (32 data + 4 pad = 144B)
#define KSTEPS (DDIM / BK)                    // 24
#define NPART ((NROWS / BM) * (NROWS / BN))   // 4096 partial sums

#define LOGIT_SCALE 2.302585092994046f
#define LOGIT_BIAS  (-10.0f)
#define LOG2E_F     1.4426950408889634f
#define LN2_F       0.6931471805599453f

// Load 16 consecutive fp32 from LDS (4x ds_load_b128), convert to a bf16
// WMMA fragment slice with a single vector fptrunc (RNE).
static __device__ __forceinline__ v16bf frag_from_lds_f32(const float* p) {
    union {
        v16f   v;
        float4 f[4];
    } u;
    const float4* fp = (const float4*)p;            // 16B-aligned LDS
    u.f[0] = fp[0];
    u.f[1] = fp[1];
    u.f[2] = fp[2];
    u.f[3] = fp[3];
    return __builtin_convertvector(u.v, v16bf);
}

static __device__ __forceinline__ unsigned lds_off(const void* p) {
    // generic AS3 cast: low 32 bits are the LDS byte offset
    return (unsigned)(unsigned long long)(size_t)p;
}

// Issue one TDM transfer: 2-D tile (tile_dim0=32 fp32 x tile_dim1=128 rows)
// from a row-major [8192 x 768] fp32 tensor into LDS, padding each 32-dword
// row with 4 dwords (-> 36-dword LDS row stride, bank-conflict free).
static __device__ __forceinline__ void tdm_load_tile(const float* gsrc,
                                                     unsigned lds_byte) {
    unsigned long long ga = (unsigned long long)(size_t)gsrc;

    v4u g0;
    g0[0] = 1u;                                        // count=1, user desc
    g0[1] = lds_byte;                                  // lds_addr
    g0[2] = (unsigned)(ga & 0xFFFFFFFFu);              // global_addr lo
    g0[3] = (unsigned)((ga >> 32) & 0x01FFFFFFu)       // global_addr hi (57b)
            | (2u << 30);                              // type=2 ("image")

    v8i g1;
    g1[0] = (int)((2u << 16)                           // data_size = 4B
                 | (1u << 20)                          // pad_enable
                 | (4u << 22)                          // pad_interval: 32 dw
                 | (3u << 25));                        // pad_amount: 4 dw
    g1[1] = (int)(((unsigned)DDIM & 0xFFFFu) << 16);   // tensor_dim0 lo16
    g1[2] = (int)((((unsigned)DDIM >> 16) & 0xFFFFu)   // tensor_dim0 hi16
                 | (((unsigned)NROWS & 0xFFFFu) << 16)); // tensor_dim1 lo16
    g1[3] = (int)((((unsigned)NROWS >> 16) & 0xFFFFu)  // tensor_dim1 hi16
                 | ((unsigned)BK << 16));              // tile_dim0 = 32
    g1[4] = (int)(unsigned)BM;                         // tile_dim1=128, dim2=0
    g1[5] = (int)(unsigned)DDIM;                       // tensor_dim0_stride lo
    g1[6] = 0;                                         // stride hi / dim1_stride
    g1[7] = 0;

    v4i z4 = (v4i)0;
#if defined(__clang_major__) && (__clang_major__ <= 22)
    // ROCm 7.2 / clang-22: 5-arg form
    __builtin_amdgcn_tensor_load_to_lds(g0, g1, z4, z4, 0);
#else
    // clang-23+ (therock-10.0 headers): 6-arg form
    v8i z8 = (v8i)0;
    __builtin_amdgcn_tensor_load_to_lds(g0, g1, z4, z4, z8, 0);
#endif
}

static __device__ __forceinline__ void wait_tensorcnt0() {
#if __has_builtin(__builtin_amdgcn_s_wait_tensorcnt)
    __builtin_amdgcn_s_wait_tensorcnt(0);
#else
    asm volatile("s_wait_tensorcnt 0x0" ::: "memory");
#endif
}

__global__ __launch_bounds__(256)
void siglip_wmma_tdm(const float* __restrict__ img,
                     const float* __restrict__ txt,
                     float* __restrict__ part) {
    __shared__ __align__(16) float As[2][BM * SKF];   // 2 x 18KB
    __shared__ __align__(16) float Bs[2][BN * SKF];   // 2 x 18KB
    __shared__ float red[256];

    const int tid = threadIdx.x;
    const int bMr = blockIdx.y * BM;    // image-row base of this block
    const int bNc = blockIdx.x * BN;    // text-row  base of this block

    // wave / lane mapping: 8 waves, 2(M) x 4(N); wave tile = 64x32
    const int wave = tid >> 5;
    const int wm   = wave >> 2;         // 0..1
    const int wn   = wave & 3;          // 0..3
    const int lane = tid & 31;
    const int lr   = lane & 15;         // row within 16-tile
    const int lk   = (lane >> 4) << 4;  // K half: 0 or 16

    const bool issuer = (wave == 0);    // one wave drives the TDM

    v8f acc[4][2] = {};                 // 8 f32 accumulator tiles (16x16)

    // prologue: DMA tile 0 of both operands into buffer 0
    if (issuer) {
        tdm_load_tile(img + (size_t)bMr * DDIM, lds_off(&As[0][0]));
        tdm_load_tile(txt + (size_t)bNc * DDIM, lds_off(&Bs[0][0]));
    }

    for (int it = 0; it < KSTEPS; ++it) {
        const int cur = it & 1;

        // tile `it` fully landed in LDS (TDM done-signal is per instruction)
        if (issuer) wait_tensorcnt0();
        __syncthreads();   // publish TDM writes; everyone done reading buf^1

        // overlap: DMA tile it+1 into the other buffer while we compute
        if (issuer && (it + 1 < KSTEPS)) {
            const size_t knext = (size_t)(it + 1) * BK;
            tdm_load_tile(img + (size_t)bMr * DDIM + knext,
                          lds_off(&As[cur ^ 1][0]));
            tdm_load_tile(txt + (size_t)bNc * DDIM + knext,
                          lds_off(&Bs[cur ^ 1][0]));
        }

        // fragments: ds_load_b128 x4 (fp32) + vector fptrunc to bf16
        v16bf fa[4], fb[2];
#pragma unroll
        for (int tm = 0; tm < 4; ++tm)
            fa[tm] = frag_from_lds_f32(
                &As[cur][(wm * 64 + tm * 16 + lr) * SKF + lk]);
#pragma unroll
        for (int tn = 0; tn < 2; ++tn)
            fb[tn] = frag_from_lds_f32(
                &Bs[cur][(wn * 32 + tn * 16 + lr) * SKF + lk]);

        // 8 WMMAs per wave per K-step
#pragma unroll
        for (int tm = 0; tm < 4; ++tm) {
#pragma unroll
            for (int tn = 0; tn < 2; ++tn) {
                acc[tm][tn] = __builtin_amdgcn_wmma_f32_16x16x32_bf16(
                    false, fa[tm], false, fb[tn],
                    (short)0, acc[tm][tn], false, false);
            }
        }
    }

    // ---- fused epilogue: softplus(-label * logit) -------------------------
    // softplus(y) = max(y,0) + ln2 * log2(1 + exp2(-|y|*log2e))
    // C/D layout: lanes 0-15: vgpr r -> M=r, N=lane; lanes 16-31: M=r+8.
    float lsum = 0.0f;
    const int rofs = (lane >> 4) << 3;
    const int cidx = lane & 15;
#pragma unroll
    for (int tm = 0; tm < 4; ++tm) {
#pragma unroll
        for (int tn = 0; tn < 2; ++tn) {
#pragma unroll
            for (int v = 0; v < 8; ++v) {
                float z = fmaf(acc[tm][tn][v], LOGIT_SCALE, LOGIT_BIAS);
                int row = bMr + wm * 64 + tm * 16 + v + rofs;
                int col = bNc + wn * 32 + tn * 16 + cidx;
                float y = (row == col) ? -z : z;       // y = -label * z
                float u = __builtin_amdgcn_exp2f(-fabsf(y) * LOG2E_F);
                lsum += fmaxf(y, 0.0f)
                      + LN2_F * __builtin_amdgcn_logf(1.0f + u);
            }
        }
    }

    // ---- deterministic block reduction ------------------------------------
    red[tid] = lsum;
    __syncthreads();
#pragma unroll
    for (int s = 128; s > 0; s >>= 1) {
        if (tid < s) red[tid] += red[tid + s];
        __syncthreads();
    }
    if (tid == 0) part[blockIdx.y * gridDim.x + blockIdx.x] = red[0];
}

__global__ __launch_bounds__(256)
void siglip_reduce_final(const float* __restrict__ part,
                         float* __restrict__ out) {
    __shared__ float red[256];
    const int tid = threadIdx.x;
    float s = 0.0f;
    for (int i = tid; i < NPART; i += 256) s += part[i];   // fixed order
    red[tid] = s;
    __syncthreads();
#pragma unroll
    for (int k = 128; k > 0; k >>= 1) {
        if (tid < k) red[tid] += red[tid + k];
        __syncthreads();
    }
    if (tid == 0) out[0] = red[0] * (1.0f / (float)NROWS);
}

extern "C" void kernel_launch(void* const* d_in, const int* in_sizes, int n_in,
                              void* d_out, int out_size, void* d_ws, size_t ws_size,
                              hipStream_t stream) {
    const float* img = (const float*)d_in[0];
    const float* txt = (const float*)d_in[1];
    float* part = (float*)d_ws;          // NPART floats of scratch
    float* out  = (float*)d_out;         // single fp32 scalar

    dim3 grid(NROWS / BN, NROWS / BM);   // 64 x 64 blocks
    siglip_wmma_tdm<<<grid, 256, 0, stream>>>(img, txt, part);
    siglip_reduce_final<<<1, 256, 0, stream>>>(part, out);
}
